// GCN_45758581572292
// MI455X (gfx1250) — matrix operand order repaired and checked by reading, not compiled, not run on hardware
//
#include <hip/hip_runtime.h>

// ---------------------------------------------------------------------------
// MI455X (gfx1250, wave32) hetero-GNN forward. Live path only (c2/t2/trk dead).
// Dense GEMMs use v_wmma_f32_16x16x32_bf16 (bf16 in, f32 accumulate).
// ---------------------------------------------------------------------------

typedef __bf16 bf16_t;
typedef __attribute__((ext_vector_type(16))) __bf16 v16bf;
typedef __attribute__((ext_vector_type(8)))  __bf16 v8bf;
typedef __attribute__((ext_vector_type(8)))  float  v8f;

#define NC    30000   // cmp nodes
#define NS    30000   // slk nodes
#define ECS   200000
#define ECC   200000
#define DC    86      // cmp feature dim
#define DSL   72      // slk feature dim
#define DPAD  96      // padded K for layer-1 (multiple of 32)
#define GNNW  256
#define HIDW  64
#define NROT  36

// ------------------------------ utility ------------------------------------

__global__ void k_zero(float* p, long n) {
  long i = (long)blockIdx.x * blockDim.x + threadIdx.x;
  if (i < n) p[i] = 0.0f;
}

// ----------------------- embedding-concat kernels ---------------------------
// one block (96 threads) per node; writes f32 row (86) + zero-padded bf16 row (96)

__global__ void k_embed_cmp(const float* __restrict__ x,
                            const float* __restrict__ et, const float* __restrict__ ed,
                            const float* __restrict__ el, const float* __restrict__ er,
                            float* __restrict__ hf, bf16_t* __restrict__ hb) {
  int i = blockIdx.x;
  int d = threadIdx.x;            // 0..95
  const float* xr = x + (long)i * 12;
  float v = 0.0f; bool valid = true;
  if (d < 13)      v = et[(int)xr[0] * 13 + d];
  else if (d < 64) v = ed[(int)xr[1] * 51 + (d - 13)];
  else if (d < 68) v = el[(int)xr[2] * 4  + (d - 64)];
  else if (d < 78) v = er[(int)xr[3] * 10 + (d - 68)];
  else if (d < 86) v = xr[4 + (d - 78)];
  else valid = false;
  if (valid) hf[(long)i * DC + d] = v;
  hb[(long)i * DPAD + d] = valid ? (bf16_t)v : (bf16_t)0.0f;
}

__global__ void k_embed_slk(const float* __restrict__ x,
                            const float* __restrict__ et, const float* __restrict__ ed,
                            bf16_t* __restrict__ hb) {
  int i = blockIdx.x;
  int d = threadIdx.x;            // 0..95
  const float* xr = x + (long)i * 10;
  float v = 0.0f; bool valid = true;
  if (d < 13)      v = et[(int)xr[0] * 13 + d];
  else if (d < 64) v = ed[(int)xr[1] * 51 + (d - 13)];
  else if (d < 72) v = xr[2 + (d - 64)];
  else valid = false;
  hb[(long)i * DPAD + d] = valid ? (bf16_t)v : (bf16_t)0.0f;
}

// --------------------------- edge aggregation ------------------------------
// blockIdx = edge, threads cover feature dims; fp32 global atomics

__global__ void k_scatter86(const float* __restrict__ h,
                            const int* __restrict__ src, const int* __restrict__ dst,
                            float* __restrict__ acc, float* __restrict__ cnt) {
  int e = blockIdx.x;
  int d = threadIdx.x;            // 0..95
  int s = src[e], t = dst[e];
  if (d < DC) atomicAdd(&acc[(long)t * DC + d], h[(long)s * DC + d]);
  if (d == DC) atomicAdd(&cnt[t], 1.0f);
}

__global__ void k_scatter256(const float* __restrict__ h,
                             const int* __restrict__ src, const int* __restrict__ dst,
                             float* __restrict__ acc) {
  int e = blockIdx.x;
  int d = threadIdx.x;            // 0..255
  int s = src[e], t = dst[e];
  atomicAdd(&acc[(long)t * GNNW + d], h[(long)s * GNNW + d]);
}

// mean = acc / max(cnt,1), write bf16 zero-padded to Dpad
__global__ void k_mean_bf(const float* __restrict__ acc, const float* __restrict__ cnt,
                          bf16_t* __restrict__ out, int n, int D, int Dpad) {
  long tid = (long)blockIdx.x * blockDim.x + threadIdx.x;
  if (tid >= (long)n * Dpad) return;
  int i = (int)(tid / Dpad);
  int d = (int)(tid - (long)i * Dpad);
  float v = 0.0f;
  if (d < D) v = acc[(long)i * D + d] / fmaxf(cnt[i], 1.0f);
  out[tid] = (bf16_t)v;
}

// weight convert + transpose: W[K x N] f32 -> Wt[N x Kpad] bf16 (zero-pad K)
__global__ void k_wt(const float* __restrict__ W, bf16_t* __restrict__ Wt,
                     int K, int N, int Kpad) {
  long tid = (long)blockIdx.x * blockDim.x + threadIdx.x;
  if (tid >= (long)N * Kpad) return;
  int n = (int)(tid / Kpad);
  int k = (int)(tid - (long)n * Kpad);
  Wt[tid] = (k < K) ? (bf16_t)W[(long)k * N + n] : (bf16_t)0.0f;
}

// ------------------------------ WMMA GEMM ----------------------------------
// Fragment layouts per CDNA5 ISA 7.12.2 (wave32):
//   A 16x32 bf16 : lane<16 -> M=lane, K = kb..kb+7 & kb+16..kb+23 (kb=0)
//                  lane>=16 -> kb=8
//   B 32x16 bf16 : N = lane&15, same K split as A (kb=0 / kb=8)
//   C/D 16x16 f32: N = lane&15, VGPR r -> M = r (+8 for lanes>=16)

__device__ __forceinline__ v16bf load_a_frag(const bf16_t* A, int lda, int m0,
                                             int kk, int lane) {
  int m  = m0 + (lane & 15);
  int kb = kk + ((lane & 16) ? 8 : 0);
  const bf16_t* p = A + (long)m * lda + kb;
  v8bf lo = *(const v8bf*)(p);
  v8bf hi = *(const v8bf*)(p + 16);
  v16bf r;
#pragma unroll
  for (int j = 0; j < 8; ++j) { r[j] = lo[j]; r[8 + j] = hi[j]; }
  return r;
}

__device__ __forceinline__ v16bf load_b_frag(const bf16_t* Bt, int ldb, int n0,
                                             int kk, int lane) {
  int n  = n0 + (lane & 15);
  int kh = kk + ((lane & 16) ? 8 : 0);
  const bf16_t* p = Bt + (long)n * ldb + kh;
  v8bf lo = *(const v8bf*)(p);
  v8bf hi = *(const v8bf*)(p + 16);
  v16bf r;
#pragma unroll
  for (int j = 0; j < 8; ++j) { r[j] = lo[j]; r[8 + j] = hi[j]; }
  return r;
}

// D = act( A1@B1t' (+ A2@B2t') (+ bias) ), per-wave 16x64 output strip.
// A row-major [M x K] bf16 (K multiple of 32); Bt row-major [N x K] bf16
// (pre-transposed weight). M multiple of 16, N multiple of 64.
template <int K1, int K2>
__global__ void k_gemm(const bf16_t* __restrict__ A1, const bf16_t* __restrict__ B1t,
                       const bf16_t* __restrict__ A2, const bf16_t* __restrict__ B2t,
                       const float* __restrict__ bias, int relu,
                       float* __restrict__ outF, bf16_t* __restrict__ outB,
                       int M, int N) {
  int lane = threadIdx.x & 31;
  int wave = threadIdx.x >> 5;
  int task = blockIdx.x * (blockDim.x >> 5) + wave;
  int nt   = N >> 6;                      // 16x64 strips per row-tile
  int tm   = task / nt;
  int tn   = (task - tm * nt) << 6;
  int m0   = tm * 16;
  if (m0 >= M) return;

  v8f c0 = {}, c1 = {}, c2 = {}, c3 = {};

#pragma unroll
  for (int kk = 0; kk < K1; kk += 32) {
    if (kk + 32 < K1)
      __builtin_prefetch((const void*)(A1 + (long)(m0 + (lane & 15)) * K1 + kk + 32), 0, 1);
    v16bf a  = load_a_frag(A1, K1, m0, kk, lane);
    v16bf b0 = load_b_frag(B1t, K1, tn +  0, kk, lane);
    v16bf b1 = load_b_frag(B1t, K1, tn + 16, kk, lane);
    v16bf b2 = load_b_frag(B1t, K1, tn + 32, kk, lane);
    v16bf b3 = load_b_frag(B1t, K1, tn + 48, kk, lane);
    c0 = __builtin_amdgcn_wmma_f32_16x16x32_bf16(false, a, false, b0, (short)0, c0, false, false);
    c1 = __builtin_amdgcn_wmma_f32_16x16x32_bf16(false, a, false, b1, (short)0, c1, false, false);
    c2 = __builtin_amdgcn_wmma_f32_16x16x32_bf16(false, a, false, b2, (short)0, c2, false, false);
    c3 = __builtin_amdgcn_wmma_f32_16x16x32_bf16(false, a, false, b3, (short)0, c3, false, false);
  }
  if constexpr (K2 > 0) {
#pragma unroll
    for (int kk = 0; kk < K2; kk += 32) {
      v16bf a  = load_a_frag(A2, K2, m0, kk, lane);
      v16bf b0 = load_b_frag(B2t, K2, tn +  0, kk, lane);
      v16bf b1 = load_b_frag(B2t, K2, tn + 16, kk, lane);
      v16bf b2 = load_b_frag(B2t, K2, tn + 32, kk, lane);
      v16bf b3 = load_b_frag(B2t, K2, tn + 48, kk, lane);
      c0 = __builtin_amdgcn_wmma_f32_16x16x32_bf16(false, a, false, b0, (short)0, c0, false, false);
      c1 = __builtin_amdgcn_wmma_f32_16x16x32_bf16(false, a, false, b1, (short)0, c1, false, false);
      c2 = __builtin_amdgcn_wmma_f32_16x16x32_bf16(false, a, false, b2, (short)0, c2, false, false);
      c3 = __builtin_amdgcn_wmma_f32_16x16x32_bf16(false, a, false, b3, (short)0, c3, false, false);
    }
  }

  int colb = tn + (lane & 15);
  int row0 = m0 + ((lane & 16) ? 8 : 0);
#pragma unroll
  for (int t = 0; t < 4; ++t) {
    v8f cc = (t == 0) ? c0 : (t == 1) ? c1 : (t == 2) ? c2 : c3;
    int col = colb + t * 16;
    float bv = bias ? bias[col] : 0.0f;
#pragma unroll
    for (int r = 0; r < 8; ++r) {
      float v = cc[r] + bv;
      if (relu) v = fmaxf(v, 0.0f);
      long off = (long)(row0 + r) * N + col;
      if (outF) outF[off] = v;
      if (outB) outB[off] = (bf16_t)v;
    }
  }
}

// ------------------------------- epilogue ----------------------------------

__global__ void k_copy2(const float* __restrict__ s, float* __restrict__ d1,
                        float* __restrict__ d2, long n) {
  long i = (long)blockIdx.x * blockDim.x + threadIdx.x;
  if (i < n) { float v = s[i]; d1[i] = v; d2[i] = v; }
}

__global__ void k_heads(const float* __restrict__ h,
                        const float* __restrict__ Wrot, const float* __restrict__ brot,
                        const float* __restrict__ Wx, const float* __restrict__ bx,
                        const float* __restrict__ Wy, const float* __restrict__ by,
                        float* __restrict__ rot, float* __restrict__ xo,
                        float* __restrict__ yo, int n) {
  long tid = (long)blockIdx.x * blockDim.x + threadIdx.x;
  int i = (int)(tid / 38);
  int j = (int)(tid - (long)i * 38);
  if (i >= n) return;
  const float* hr = h + (long)i * HIDW;
  float s = 0.0f;
  if (j < NROT) {
#pragma unroll
    for (int k = 0; k < HIDW; ++k) s += hr[k] * Wrot[k * NROT + j];
    rot[(long)i * NROT + j] = s + brot[j];
  } else if (j == NROT) {
#pragma unroll
    for (int k = 0; k < HIDW; ++k) s += hr[k] * Wx[k];
    xo[i] = s + bx[0];
  } else {
#pragma unroll
    for (int k = 0; k < HIDW; ++k) s += hr[k] * Wy[k];
    yo[i] = s + by[0];
  }
}

// ------------------------------- launcher ----------------------------------

extern "C" void kernel_launch(void* const* d_in, const int* in_sizes, int n_in,
                              void* d_out, int out_size, void* d_ws, size_t ws_size,
                              hipStream_t stream) {
  const float* x_cmp    = (const float*)d_in[0];
  const float* x_slk    = (const float*)d_in[1];
  const float* emb_tool = (const float*)d_in[3];
  const float* emb_des  = (const float*)d_in[4];
  const float* emb_lay  = (const float*)d_in[5];
  const float* emb_rot  = (const float*)d_in[6];
  const float* Wl1_cs   = (const float*)d_in[7];   // [86 x 256]
  const float* Wr1_cs   = (const float*)d_in[8];   // [72 x 256]
  const float* Wl2_cs   = (const float*)d_in[9];   // [256 x 256]
  const float* Wr2_cs   = (const float*)d_in[10];  // [256 x 256]
  const float* Wl1_cc   = (const float*)d_in[11];  // [86 x 256]
  const float* Wr1_cc   = (const float*)d_in[12];  // [86 x 256]
  const float* W_in     = (const float*)d_in[23];  // [256 x 64]
  const float* b_in     = (const float*)d_in[24];
  const float* W_lin    = (const float*)d_in[25];  // [64 x 64]
  const float* b_lin    = (const float*)d_in[26];
  const float* W_rot    = (const float*)d_in[27];  // [64 x 36]
  const float* b_rot    = (const float*)d_in[28];
  const float* W_x      = (const float*)d_in[29];
  const float* b_x      = (const float*)d_in[30];
  const float* W_y      = (const float*)d_in[31];
  const float* b_y      = (const float*)d_in[32];
  const int* e_cs_src   = (const int*)d_in[33];
  const int* e_cs_dst   = (const int*)d_in[34];
  const int* e_cc_src   = (const int*)d_in[35];
  const int* e_cc_dst   = (const int*)d_in[36];

  char* ws = (char*)d_ws;
  auto up = [](size_t x) { return (x + 255) & ~(size_t)255; };

  // region A (reused by acc2 in layer 2)
  size_t o_hcmp_f = 0;
  size_t o_acc_cs = up(o_hcmp_f + (size_t)NC * DC * 4);
  size_t o_acc_cc = up(o_acc_cs + (size_t)NS * DC * 4);
  size_t regA_end = up(o_acc_cc + (size_t)NC * DC * 4);
  size_t o_acc2   = 0;                       // NS*256*4 = 30.72MB <= regA_end
  size_t o = regA_end;
  size_t o_cnt_cs = o; o = up(o + (size_t)NS * 4);
  size_t o_cnt_cc = o; o = up(o + (size_t)NC * 4);
  size_t o_hcmp_b = o; o = up(o + (size_t)NC * DPAD * 2);
  size_t o_hslk_b = o; o = up(o + (size_t)NS * DPAD * 2);
  size_t o_mcs_b  = o; o = up(o + (size_t)NS * DPAD * 2);
  size_t o_mcc_b  = o; o = up(o + (size_t)NC * DPAD * 2);
  size_t o_c1_f   = o; o = up(o + (size_t)NC * GNNW * 4);
  size_t o_s1_b   = o; o = up(o + (size_t)NS * GNNW * 2);
  size_t o_m2_b   = o; o = up(o + (size_t)NS * GNNW * 2);
  size_t o_s2_b   = o; o = up(o + (size_t)NS * GNNW * 2);
  size_t o_h1_b   = o; o = up(o + (size_t)NS * HIDW * 2);
  size_t o_wl1cs  = o; o = up(o + (size_t)GNNW * DPAD * 2);
  size_t o_wr1cs  = o; o = up(o + (size_t)GNNW * DPAD * 2);
  size_t o_wl1cc  = o; o = up(o + (size_t)GNNW * DPAD * 2);
  size_t o_wr1cc  = o; o = up(o + (size_t)GNNW * DPAD * 2);
  size_t o_wl2cs  = o; o = up(o + (size_t)GNNW * GNNW * 2);
  size_t o_wr2cs  = o; o = up(o + (size_t)GNNW * GNNW * 2);
  size_t o_win    = o; o = up(o + (size_t)HIDW * GNNW * 2);
  size_t o_wlin   = o; o = up(o + (size_t)HIDW * HIDW * 2);
  (void)ws_size; (void)in_sizes; (void)n_in; (void)out_size;

  float*  hcmp_f = (float*)(ws + o_hcmp_f);
  float*  acc_cs = (float*)(ws + o_acc_cs);
  float*  acc_cc = (float*)(ws + o_acc_cc);
  float*  acc2   = (float*)(ws + o_acc2);
  float*  cnt_cs = (float*)(ws + o_cnt_cs);
  float*  cnt_cc = (float*)(ws + o_cnt_cc);
  bf16_t* hcmp_b = (bf16_t*)(ws + o_hcmp_b);
  bf16_t* hslk_b = (bf16_t*)(ws + o_hslk_b);
  bf16_t* mcs_b  = (bf16_t*)(ws + o_mcs_b);
  bf16_t* mcc_b  = (bf16_t*)(ws + o_mcc_b);
  float*  c1_f   = (float*)(ws + o_c1_f);
  bf16_t* s1_b   = (bf16_t*)(ws + o_s1_b);
  bf16_t* m2_b   = (bf16_t*)(ws + o_m2_b);
  bf16_t* s2_b   = (bf16_t*)(ws + o_s2_b);
  bf16_t* h1_b   = (bf16_t*)(ws + o_h1_b);
  bf16_t* wl1cs  = (bf16_t*)(ws + o_wl1cs);
  bf16_t* wr1cs  = (bf16_t*)(ws + o_wr1cs);
  bf16_t* wl1cc  = (bf16_t*)(ws + o_wl1cc);
  bf16_t* wr1cc  = (bf16_t*)(ws + o_wr1cc);
  bf16_t* wl2cs  = (bf16_t*)(ws + o_wl2cs);
  bf16_t* wr2cs  = (bf16_t*)(ws + o_wr2cs);
  bf16_t* win_t  = (bf16_t*)(ws + o_win);
  bf16_t* wlin_t = (bf16_t*)(ws + o_wlin);

  float* out  = (float*)d_out;
  const long H_SZ = (long)NS * HIDW;                 // 1,920,000
  float* out_h   = out;
  float* out_rot = out + 3 * H_SZ;
  float* out_x   = out + 3 * H_SZ + (long)NS * NROT;
  float* out_y   = out_x + NS;

  auto zeros = [&](float* p, long n) {
    k_zero<<<dim3((unsigned)((n + 255) / 256)), dim3(256), 0, stream>>>(p, n);
  };

  // 1) embeddings
  k_embed_cmp<<<dim3(NC), dim3(DPAD), 0, stream>>>(x_cmp, emb_tool, emb_des, emb_lay,
                                                   emb_rot, hcmp_f, hcmp_b);
  k_embed_slk<<<dim3(NS), dim3(DPAD), 0, stream>>>(x_slk, emb_tool, emb_des, hslk_b);

  // 2) weight convert+transpose (bf16, padded K)
  auto wt = [&](const float* W, bf16_t* Wt, int K, int N, int Kpad) {
    long n = (long)N * Kpad;
    k_wt<<<dim3((unsigned)((n + 255) / 256)), dim3(256), 0, stream>>>(W, Wt, K, N, Kpad);
  };
  wt(Wl1_cs, wl1cs, DC,  GNNW, DPAD);
  wt(Wr1_cs, wr1cs, DSL, GNNW, DPAD);
  wt(Wl1_cc, wl1cc, DC,  GNNW, DPAD);
  wt(Wr1_cc, wr1cc, DC,  GNNW, DPAD);
  wt(Wl2_cs, wl2cs, GNNW, GNNW, GNNW);
  wt(Wr2_cs, wr2cs, GNNW, GNNW, GNNW);
  wt(W_in,   win_t, GNNW, HIDW, GNNW);
  wt(W_lin,  wlin_t, HIDW, HIDW, HIDW);

  // 3) layer-1 mean aggregation (cc->cmp, cs->slk)
  zeros(acc_cc, (long)NC * DC); zeros(cnt_cc, NC);
  zeros(acc_cs, (long)NS * DC); zeros(cnt_cs, NS);
  k_scatter86<<<dim3(ECC), dim3(DPAD), 0, stream>>>(hcmp_f, e_cc_src, e_cc_dst, acc_cc, cnt_cc);
  k_scatter86<<<dim3(ECS), dim3(DPAD), 0, stream>>>(hcmp_f, e_cs_src, e_cs_dst, acc_cs, cnt_cs);
  {
    long n = (long)NC * DPAD;
    k_mean_bf<<<dim3((unsigned)((n + 255) / 256)), dim3(256), 0, stream>>>(acc_cc, cnt_cc, mcc_b, NC, DC, DPAD);
    k_mean_bf<<<dim3((unsigned)((n + 255) / 256)), dim3(256), 0, stream>>>(acc_cs, cnt_cs, mcs_b, NS, DC, DPAD);
  }

  // 4) layer-1 GEMMs: c1 (f32, feeds layer-2 scatter), s1 (bf16 root operand)
  {
    int tasks = (NC / 16) * (GNNW / 64);
    unsigned blocks = (unsigned)((tasks + 7) / 8);
    k_gemm<DPAD, DPAD><<<dim3(blocks), dim3(256), 0, stream>>>(
        mcc_b, wl1cc, hcmp_b, wr1cc, nullptr, 1, c1_f, nullptr, NC, GNNW);
    k_gemm<DPAD, DPAD><<<dim3(blocks), dim3(256), 0, stream>>>(
        mcs_b, wl1cs, hslk_b, wr1cs, nullptr, 1, nullptr, s1_b, NS, GNNW);
  }

  // 5) layer-2 mean over cs edges of c1 (counts identical to layer-1 cs counts)
  zeros(acc2, (long)NS * GNNW);
  k_scatter256<<<dim3(ECS), dim3(GNNW), 0, stream>>>(c1_f, e_cs_src, e_cs_dst, acc2);
  {
    long n = (long)NS * GNNW;
    k_mean_bf<<<dim3((unsigned)((n + 255) / 256)), dim3(256), 0, stream>>>(acc2, cnt_cs, m2_b, NS, GNNW, GNNW);
  }

  // 6) s2 = relu(mean2 @ Wl2_cs + s1 @ Wr2_cs)
  {
    int tasks = (NS / 16) * (GNNW / 64);
    unsigned blocks = (unsigned)((tasks + 7) / 8);
    k_gemm<GNNW, GNNW><<<dim3(blocks), dim3(256), 0, stream>>>(
        m2_b, wl2cs, s1_b, wr2cs, nullptr, 1, nullptr, s2_b, NS, GNNW);
  }

  // 7) MLP: h1 = relu(s2@W_in+b_in); h = relu(h1@W_lin+b_lin) -> d_out
  {
    int tasks = (NS / 16) * (HIDW / 64);
    unsigned blocks = (unsigned)((tasks + 7) / 8);
    k_gemm<GNNW, 0><<<dim3(blocks), dim3(256), 0, stream>>>(
        s2_b, win_t, nullptr, nullptr, b_in, 1, nullptr, h1_b, NS, HIDW);
    k_gemm<HIDW, 0><<<dim3(blocks), dim3(256), 0, stream>>>(
        h1_b, wlin_t, nullptr, nullptr, b_lin, 1, out_h, nullptr, NS, HIDW);
  }

  // 8) replicate h x3 and heads
  k_copy2<<<dim3((unsigned)((H_SZ + 255) / 256)), dim3(256), 0, stream>>>(
      out_h, out_h + H_SZ, out_h + 2 * H_SZ, H_SZ);
  {
    long n = (long)NS * 38;
    k_heads<<<dim3((unsigned)((n + 255) / 256)), dim3(256), 0, stream>>>(
        out_h, W_rot, b_rot, W_x, b_x, W_y, b_y, out_rot, out_x, out_y, NS);
  }
}
